// Llama4TextExpertsLinear_8916352106544
// MI455X (gfx1250) — matrix operand order repaired and checked by reading, not compiled, not run on hardware
//
#include <hip/hip_runtime.h>
#include <hip/hip_bf16.h>

#define E_ 16
#define T_ 512
#define H_ 2048
#define I_ 4096

// CDNA5 WMMA operand types
typedef __attribute__((ext_vector_type(16))) __bf16 v16bf;
typedef __attribute__((ext_vector_type(2)))  __bf16 v2bf;
typedef __attribute__((ext_vector_type(8)))  float  v8f;

union FragU {
    uint4 u[2];
    v16bf v;
};
static_assert(sizeof(v16bf) == 32, "v16bf must be 32B");

// fp32 -> bf16 (RNE) via native conversion; packed cvt builtin when present.
__device__ __forceinline__ unsigned int pk2bf(float lo, float hi) {
#if defined(__has_builtin) && __has_builtin(__builtin_amdgcn_cvt_pk_bf16_f32)
    union { v2bf v; unsigned int u; } c;
    c.v = __builtin_amdgcn_cvt_pk_bf16_f32(lo, hi);
    return c.u;
#else
    union { v2bf v; unsigned int u; } c;
    c.v.x = (__bf16)lo;   // clang fptrunc -> native v_cvt on gfx1250
    c.v.y = (__bf16)hi;
    return c.u;
#endif
}

__device__ __forceinline__ unsigned short f2bf(float f) {
    union { __bf16 b; unsigned short s; } c;
    c.b = (__bf16)f;
    return c.s;
}

// ---------------------------------------------------------------------------
// Kernel 1: act[e,t,i] = up * silu(gate), gate/up = X . W1^T  (K = H = 2048)
// grid (T/64, I/64, E), block 128 (4 waves), wave -> 32x32 of gate AND up
// BK = 64 per LDS stage (2 WMMA K-steps per barrier)
// ---------------------------------------------------------------------------
__global__ __launch_bounds__(128)
void gateup_silu_kernel(const float* __restrict__ X,
                        const float* __restrict__ W,      // (E, 2I, H)
                        unsigned short* __restrict__ act) // (E, T, I) bf16
{
    __shared__ unsigned short sA[64 * 72];
    __shared__ unsigned short sG[64 * 72];
    __shared__ unsigned short sU[64 * 72];

    const int tid  = threadIdx.x;
    const int lane = tid & 31;
    const int wv   = tid >> 5;
    const int e    = blockIdx.z;
    const int tm   = blockIdx.x;   // token tile (64 rows)
    const int tn   = blockIdx.y;   // intermediate tile (64 cols)

    const int wm = (wv & 1) * 32;
    const int wn = (wv >> 1) * 32;
    const int lr = lane & 15;      // row within 16
    const int lh = lane >> 4;      // lane half

    const size_t xBase = ((size_t)e * T_ + (size_t)tm * 64) * H_;
    const size_t gBase = ((size_t)e * 2 * I_ + (size_t)tn * 64) * (size_t)H_;
    const size_t uBase = gBase + (size_t)I_ * H_;

    v8f accg[2][2] = {};
    v8f accu[2][2] = {};

    for (int kt = 0; kt < H_ / 64; ++kt) {
        const int k0 = kt * 64;

        // ---- stage 64x64 tiles into LDS (fp32 -> bf16, packed cvt) ----
        #pragma unroll
        for (int it = 0; it < 8; ++it) {
            const int j   = tid + it * 128;   // 0..1023 float4 chunks
            const int row = j >> 4;
            const int c4  = (j & 15) * 4;
            const size_t ro = (size_t)row * H_ + k0 + c4;
            float4 a = *(const float4*)(X + xBase + ro);
            float4 g = *(const float4*)(W + gBase + ro);
            float4 u = *(const float4*)(W + uBase + ro);
            *(uint2*)&sA[row * 72 + c4] = make_uint2(pk2bf(a.x, a.y), pk2bf(a.z, a.w));
            *(uint2*)&sG[row * 72 + c4] = make_uint2(pk2bf(g.x, g.y), pk2bf(g.z, g.w));
            *(uint2*)&sU[row * 72 + c4] = make_uint2(pk2bf(u.x, u.y), pk2bf(u.z, u.w));
        }
        // prefetch next K tile of weights (global_prefetch_b8 on gfx1250)
        if (kt + 1 < H_ / 64) {
            __builtin_prefetch(W + gBase + (size_t)(tid & 63) * H_ + k0 + 64, 0, 1);
            __builtin_prefetch(W + uBase + (size_t)(tid & 63) * H_ + k0 + 64, 0, 1);
        }
        __syncthreads();

        // ---- two WMMA K-steps per stage ----
        #pragma unroll
        for (int s = 0; s < 2; ++s) {
            const int kk = s * 32;
            FragU af[2], gf[2], uf[2];
            #pragma unroll
            for (int mi = 0; mi < 2; ++mi) {
                // A: lane-half q holds K in {16q..16q+15}, contiguous
                const unsigned short* p = &sA[(wm + mi * 16 + lr) * 72 + kk + lh * 16];
                af[mi].u[0] = *(const uint4*)(p);
                af[mi].u[1] = *(const uint4*)(p + 8);
            }
            #pragma unroll
            for (int ni = 0; ni < 2; ++ni) {
                // B: lane-half q holds K in {16q..16q+15}, contiguous
                const unsigned short* pg = &sG[(wn + ni * 16 + lr) * 72 + kk + lh * 16];
                gf[ni].u[0] = *(const uint4*)(pg);
                gf[ni].u[1] = *(const uint4*)(pg + 8);
                const unsigned short* pu = &sU[(wn + ni * 16 + lr) * 72 + kk + lh * 16];
                uf[ni].u[0] = *(const uint4*)(pu);
                uf[ni].u[1] = *(const uint4*)(pu + 8);
            }
            #pragma unroll
            for (int mi = 0; mi < 2; ++mi)
                #pragma unroll
                for (int ni = 0; ni < 2; ++ni) {
                    accg[mi][ni] = __builtin_amdgcn_wmma_f32_16x16x32_bf16(
                        false, af[mi].v, false, gf[ni].v, (short)0, accg[mi][ni],
                        false, false);
                    accu[mi][ni] = __builtin_amdgcn_wmma_f32_16x16x32_bf16(
                        false, af[mi].v, false, uf[ni].v, (short)0, accu[mi][ni],
                        false, false);
                }
        }
        __syncthreads();
    }

    // ---- epilogue: act = up * silu(gate), bf16 store ----
    const size_t actBase = (size_t)e * T_ * I_;
    #pragma unroll
    for (int mi = 0; mi < 2; ++mi)
        #pragma unroll
        for (int ni = 0; ni < 2; ++ni) {
            const int col = tn * 64 + wn + ni * 16 + lr;
            #pragma unroll
            for (int v = 0; v < 8; ++v) {
                const int row = tm * 64 + wm + mi * 16 + lh * 8 + v;
                float g = accg[mi][ni][v];
                float s = g / (1.0f + __expf(-g));   // silu(g)
                float a = accu[mi][ni][v] * s;
                act[actBase + (size_t)row * I_ + col] = f2bf(a);
            }
        }
}

// ---------------------------------------------------------------------------
// Kernel 2: out[e,t,h] = act . W2^T  (K = I = 4096)
// grid (T/64, H/64, E), block 128 (4 waves), wave -> 32x32
// A tile (bf16, no conversion) staged via GLOBAL_LOAD_ASYNC_TO_LDS_B128
// ---------------------------------------------------------------------------
__global__ __launch_bounds__(128)
void down_kernel(const unsigned short* __restrict__ act, // (E, T, I) bf16
                 const float* __restrict__ W2,           // (E, H, I)
                 float* __restrict__ out)                 // (E*T, H)
{
    __shared__ unsigned short sA[64 * 72];
    __shared__ unsigned short sB[64 * 72];

    const int tid  = threadIdx.x;
    const int lane = tid & 31;
    const int wv   = tid >> 5;
    const int e    = blockIdx.z;
    const int tm   = blockIdx.x;   // token tile
    const int tn   = blockIdx.y;   // hidden tile

    const int wm = (wv & 1) * 32;
    const int wn = (wv >> 1) * 32;
    const int lr = lane & 15;
    const int lh = lane >> 4;

    const size_t aBase = ((size_t)e * T_ + (size_t)tm * 64) * (size_t)I_;
    const size_t bBase = ((size_t)e * H_ + (size_t)tn * 64) * (size_t)I_;

    v8f acc[2][2] = {};

    for (int kt = 0; kt < I_ / 64; ++kt) {
        const int k0 = kt * 64;

        // A tile: already bf16 -> async DMA straight into LDS (no VGPR hop).
        // Generic LDS pointer low 32 bits == LDS byte address (flat aperture rule).
        #pragma unroll
        for (int it = 0; it < 4; ++it) {
            const int j   = tid + it * 128;  // 0..511 16B chunks
            const int row = j >> 3;
            const int c8  = (j & 7) * 8;
            const unsigned ldsOff =
                (unsigned)(unsigned long long)&sA[row * 72 + c8];
            const unsigned short* gp = act + aBase + (size_t)row * I_ + k0 + c8;
            asm volatile("global_load_async_to_lds_b128 %0, %1, off"
                         :: "v"(ldsOff), "v"(gp) : "memory");
        }
        // B tile: fp32 weights -> bf16 (packed cvt)
        #pragma unroll
        for (int it = 0; it < 8; ++it) {
            const int j   = tid + it * 128;
            const int row = j >> 4;
            const int c4  = (j & 15) * 4;
            float4 b = *(const float4*)(W2 + bBase + (size_t)row * I_ + k0 + c4);
            *(uint2*)&sB[row * 72 + c4] = make_uint2(pk2bf(b.x, b.y), pk2bf(b.z, b.w));
        }
        if (kt + 1 < I_ / 64) {
            __builtin_prefetch(W2 + bBase + (size_t)(tid & 63) * I_ + k0 + 64, 0, 1);
        }
        asm volatile("s_wait_asynccnt 0x0" ::: "memory");
        __syncthreads();

        #pragma unroll
        for (int s = 0; s < 2; ++s) {
            const int kk = s * 32;
            FragU af[2], bf[2];
            #pragma unroll
            for (int mi = 0; mi < 2; ++mi) {
                const unsigned short* p = &sA[(wm + mi * 16 + lr) * 72 + kk + lh * 16];
                af[mi].u[0] = *(const uint4*)(p);
                af[mi].u[1] = *(const uint4*)(p + 8);
            }
            #pragma unroll
            for (int ni = 0; ni < 2; ++ni) {
                const unsigned short* p = &sB[(wn + ni * 16 + lr) * 72 + kk + lh * 16];
                bf[ni].u[0] = *(const uint4*)(p);
                bf[ni].u[1] = *(const uint4*)(p + 8);
            }
            #pragma unroll
            for (int mi = 0; mi < 2; ++mi)
                #pragma unroll
                for (int ni = 0; ni < 2; ++ni)
                    acc[mi][ni] = __builtin_amdgcn_wmma_f32_16x16x32_bf16(
                        false, af[mi].v, false, bf[ni].v, (short)0, acc[mi][ni],
                        false, false);
        }
        __syncthreads();
    }

    // ---- epilogue: fp32 output, row-major (E*T, H) ----
    #pragma unroll
    for (int mi = 0; mi < 2; ++mi)
        #pragma unroll
        for (int ni = 0; ni < 2; ++ni) {
            const int col = tn * 64 + wn + ni * 16 + lr;
            #pragma unroll
            for (int v = 0; v < 8; ++v) {
                const int row = tm * 64 + wm + mi * 16 + lh * 8 + v;
                out[((size_t)e * T_ + row) * H_ + col] = acc[mi][ni][v];
            }
        }
}

// ---------------------------------------------------------------------------
extern "C" void kernel_launch(void* const* d_in, const int* in_sizes, int n_in,
                              void* d_out, int out_size, void* d_ws, size_t ws_size,
                              hipStream_t stream) {
    const float* hidden = (const float*)d_in[0];          // (E*T, H)
    const float* gup_w  = (const float*)d_in[1];          // (E, 2I, H)
    const float* down_w = (const float*)d_in[2];          // (E, H, I)
    unsigned short* actw = (unsigned short*)d_ws;         // (E, T, I) bf16, 64 MB
    float* outp = (float*)d_out;

    (void)in_sizes; (void)n_in; (void)out_size; (void)ws_size;

    dim3 g1(T_ / 64, I_ / 64, E_);   // (8, 64, 16)
    gateup_silu_kernel<<<g1, 128, 0, stream>>>(hidden, gup_w, actw);

    dim3 g2(T_ / 64, H_ / 64, E_);   // (8, 32, 16)
    down_kernel<<<g2, 128, 0, stream>>>(actw, down_w, outp);
}
